// CharCNNEmbedding_79087527788780
// MI455X (gfx1250) — compile-verified, matrix operand order checked by reading, not run on hardware
//
#include <hip/hip_runtime.h>
#include <hip/hip_bf16.h>

typedef __attribute__((ext_vector_type(16))) _Float16 v16h;
typedef __attribute__((ext_vector_type(8)))  _Float16 v8h;
typedef __attribute__((ext_vector_type(2)))  _Float16 v2h;
typedef __attribute__((ext_vector_type(8)))  float    v8f;

// Fragment loader: each lane grabs its 16 f16 K-values for one row of a
// 16x32 (f16) WMMA A/B tile from LDS, per the CDNA5 16-bit A/B layout:
//   lanes 0-15 : K = base+0..7  and base+16..23
//   lanes 16-31: K = base+8..15 and base+24..31
// Row stride and base are 16B-aligned so these lower to 2x ds_load_b128.
__device__ __forceinline__ v16h ldfrag(const _Float16* base, int row,
                                       int rowStrideHalves, int ks, int lane) {
  int kbase = ks * 32 + ((lane & 16) >> 1);     // +0 or +8 halves
  const _Float16* p = base + row * rowStrideHalves + kbase;
  union { v16h v; v8h h[2]; } u;
  u.h[0] = *(const v8h*)(p);                    // K .. K+7
  u.h[1] = *(const v8h*)(p + 16);               // K+16 .. K+23
  return u.v;
}

// One conv branch: im2col-free GEMM + bias + relu + masked max over positions.
// Xw: (40 rows x 64 halves) LDS tile; im2col row p = contiguous slice at X row p.
// Bt: (NTILES*16 rows x KSIZE*64 halves) LDS tile, Bt[f][k*64+c] = w[f][c][k].
// A fragments are register-cached per M-tile and reused across all N-tiles,
// cutting LDS reads per WMMA from 4x b128 to ~2.6x b128.
template<int KSIZE, int F, int NTILES, int PVALID>
__device__ __forceinline__ void conv_branch(const _Float16* Xw, const _Float16* Bt,
                                            const float* __restrict__ bias,
                                            float* featOut, int lane) {
  constexpr int Ktot = KSIZE * 64;
  constexpr int KSTEPS = Ktot / 32;
  const int fl = lane & 15;
  const int hi = (lane & 16) >> 1;              // 0 or 8: M offset of this half-wave

  float bf[NTILES];
  float m[NTILES];
#pragma unroll
  for (int nt = 0; nt < NTILES; ++nt) {
    const int f = nt * 16 + fl;
    bf[nt] = (f < F) ? bias[f] : 0.0f;
    m[nt] = 0.0f;
  }

#pragma unroll
  for (int mt = 0; mt < 2; ++mt) {
    // Register-cache all A fragments for this M-tile (KSTEPS x 8 VGPRs).
    v16h aF[KSTEPS];
#pragma unroll
    for (int ks = 0; ks < KSTEPS; ++ks)
      aF[ks] = ldfrag(Xw, mt * 16 + fl, 64, ks, lane);

#pragma unroll
    for (int nt = 0; nt < NTILES; ++nt) {
      v8f acc = {};
#pragma unroll
      for (int ks = 0; ks < KSTEPS; ++ks) {
        v16h b = ldfrag(Bt, nt * 16 + fl, Ktot, ks, lane);
        acc = __builtin_amdgcn_wmma_f32_16x16x32_f16(
            /*neg_a=*/false, aF[ks], /*neg_b=*/false, b,
            /*c_mod=*/(short)0, acc, /*reuse_a=*/false, /*reuse_b=*/false);
      }
      // C/D layout: VGPR v, lane l -> M = v + 8*(l/16), N = l%16
#pragma unroll
      for (int v = 0; v < 8; ++v) {
        const int p = mt * 16 + hi + v;
        float y = fmaxf(acc[v] + bf[nt], 0.0f);
        m[nt] = (p < PVALID) ? fmaxf(m[nt], y) : m[nt];  // mask padded positions
      }
    }
  }

#pragma unroll
  for (int nt = 0; nt < NTILES; ++nt) {
    float mm = fmaxf(m[nt], __shfl_xor(m[nt], 16));  // combine half-wave M groups
    const int f = nt * 16 + fl;
    if (lane < 16 && f < F) featOut[f] = mm;
  }
}

__global__ __launch_bounds__(256, 1)
void charcnn_wmma_kernel(const int* __restrict__ word_idxs,
                         const int* __restrict__ char_table,
                         const float* __restrict__ char_emb,
                         const float* __restrict__ w1, const float* __restrict__ b1,
                         const float* __restrict__ w3, const float* __restrict__ b3,
                         const float* __restrict__ w5, const float* __restrict__ b5,
                         const float* __restrict__ lin_w, const float* __restrict__ lin_b,
                         float* __restrict__ out, int nWords) {
  // f16 weight tiles, transposed + zero-padded: Bt[f][k*64+c] = w[f][c][k]
  __shared__ __align__(16) _Float16 sBt1[32 * 64];     //  4 KB
  __shared__ __align__(16) _Float16 sBt3[48 * 192];    // 18 KB
  __shared__ __align__(16) _Float16 sBt5[64 * 320];    // 40 KB
  __shared__ __align__(16) _Float16 sX[8][40 * 64];    // 40 KB (per-wave X tile)
  __shared__ float sFeat[8][128];                      //  4 KB

  const int tid = threadIdx.x;

  // ---- stage weights (f32 global -> f16 LDS, transposed, zero-padded) ----
  for (int i = tid; i < 32 * 64; i += 256) {
    int f = i >> 6, c = i & 63;
    float v = (f < 30 && c < 50) ? w1[f * 50 + c] : 0.0f;       // (30,50,1)
    sBt1[i] = (_Float16)v;
  }
  for (int i = tid; i < 48 * 192; i += 256) {
    int f = i / 192, j = i % 192;
    int kk = j >> 6, c = j & 63;
    float v = (f < 40 && c < 50) ? w3[(f * 50 + c) * 3 + kk] : 0.0f;  // (40,50,3)
    sBt3[i] = (_Float16)v;
  }
  for (int i = tid; i < 64 * 320; i += 256) {
    int f = i / 320, j = i % 320;
    int kk = j >> 6, c = j & 63;
    float v = (f < 50 && c < 50) ? w5[(f * 50 + c) * 5 + kk] : 0.0f;  // (50,50,5)
    sBt5[i] = (_Float16)v;
  }
  __syncthreads();

  const int wave = tid >> 5;
  const int lane = tid & 31;
  const int n = blockIdx.x * 8 + wave;                 // one word per wave
  if (n < nWords) {
    _Float16* Xw = sX[wave];

    // ---- stage this word's char embeddings: X[l][c], C padded to 64 ----
    const int widx = word_idxs[n];
    const int ch = char_table[widx * 32 + lane];       // lane = char position
    const float* er = char_emb + (size_t)ch * 50;
#pragma unroll
    for (int c = 0; c < 50; c += 2) {
      float2 t = *(const float2*)(er + c);
      v2h h; h.x = (_Float16)t.x; h.y = (_Float16)t.y;
      *(v2h*)(Xw + lane * 64 + c) = h;
    }
#pragma unroll
    for (int c = 50; c < 64; c += 2) {                 // pad columns 50..63
      v2h z = {}; *(v2h*)(Xw + lane * 64 + c) = z;
    }
#pragma unroll
    for (int i = 0; i < 8; ++i) {                      // pad rows 32..39
      v2h z = {}; *(v2h*)(Xw + 32 * 64 + lane * 16 + i * 2) = z;
    }
    // Same-wave LDS ops are in-order: stores above are visible to the
    // fragment loads below without a workgroup barrier.

    // ---- three conv branches via WMMA; features packed [0,30)[30,70)[70,120) ----
    float* feat = sFeat[wave];
    conv_branch<1, 30, 2, 32>(Xw, sBt1, b1, feat + 0,  lane);
    conv_branch<3, 40, 3, 30>(Xw, sBt3, b3, feat + 30, lane);
    conv_branch<5, 50, 4, 28>(Xw, sBt5, b5, feat + 70, lane);

    // ---- final 120 -> 50 linear (2% of FLOPs, plain VALU f32) ----
    for (int e = lane; e < 50; e += 32) {
      float s = lin_b[e];
      const float* lw = lin_w + e * 120;
#pragma unroll 8
      for (int f = 0; f < 120; ++f) s = fmaf(feat[f], lw[f], s);
      out[(size_t)n * 50 + e] = s;
    }
  }
}

extern "C" void kernel_launch(void* const* d_in, const int* in_sizes, int n_in,
                              void* d_out, int out_size, void* d_ws, size_t ws_size,
                              hipStream_t stream) {
  (void)n_in; (void)d_ws; (void)ws_size; (void)out_size;
  const int*   word_idxs  = (const int*)d_in[0];   // (64,256) i32
  const int*   char_table = (const int*)d_in[1];   // (50000,32) i32
  const float* char_emb   = (const float*)d_in[2]; // (128,50) f32
  const float* w1 = (const float*)d_in[3];
  const float* b1 = (const float*)d_in[4];
  const float* w3 = (const float*)d_in[5];
  const float* b3 = (const float*)d_in[6];
  const float* w5 = (const float*)d_in[7];
  const float* b5 = (const float*)d_in[8];
  const float* lin_w = (const float*)d_in[9];      // (50,120) f32
  const float* lin_b = (const float*)d_in[10];     // (50,) f32
  float* out = (float*)d_out;                      // (64,256,50) f32

  const int nWords = in_sizes[0];                  // 16384
  const int grid = (nWords + 7) / 8;               // 8 waves/block, 1 word/wave
  charcnn_wmma_kernel<<<grid, 256, 0, stream>>>(
      word_idxs, char_table, char_emb, w1, b1, w3, b3, w5, b5,
      lin_w, lin_b, out, nWords);
}
